// LocalConv2D_34144990003610
// MI455X (gfx1250) — matrix operand order repaired and checked by reading, not compiled, not run on hardware
//
#include <hip/hip_runtime.h>
#include <math.h>

typedef __attribute__((ext_vector_type(16))) _Float16 v16h;
typedef __attribute__((ext_vector_type(8)))  _Float16 v8h;
typedef __attribute__((ext_vector_type(8)))  float    v8f;

#define NIMG   32
#define CIN    64
#define HH     56
#define WW     56
#define FOUT   128
#define SW     4
#define WBLK   14                  // WW / SW
#define KDIM   (CIN * 9)           // 576
#define MROWS  (NIMG * HH * WBLK)  // 25088
#define MTILE  128
#define KTILE  32
#define LDA    40                  // LDS row stride in halfs (80B, 16B-aligned)
#define XP     58                  // padded H/W
#define XPC    (XP * XP)           // 3364
#define NPAD   (NIMG * CIN * XPC)  // 6,889,472 f16 elements

// ---------------- prep 1: pad + convert input to f16 ----------------

__global__ void lc_pad_kernel(const float* __restrict__ x,
                              _Float16* __restrict__ xp) {
    int i = blockIdx.x * 256 + threadIdx.x;
    if (i >= NPAD) return;
    int iw = i % XP;
    int t  = i / XP;
    int ih = t % XP;
    int nc = t / XP;
    float v = 0.0f;
    if (ih >= 1 && ih <= HH && iw >= 1 && iw <= WW)
        v = x[(nc * HH + (ih - 1)) * WW + (iw - 1)];
    xp[i] = (_Float16)v;
}

// ---------------- prep 2: weight fake-quant ----------------

__global__ void lc_absmax_kernel(const float* __restrict__ w,
                                 float* __restrict__ out, int n) {
    __shared__ float red[256];
    int tid = threadIdx.x;
    float m = 0.0f;
    for (int i = blockIdx.x * 256 + tid; i < n; i += gridDim.x * 256)
        m = fmaxf(m, fabsf(w[i]));
    red[tid] = m;
    __syncthreads();
    for (int s = 128; s > 0; s >>= 1) {
        if (tid < s) red[tid] = fmaxf(red[tid], red[tid + s]);
        __syncthreads();
    }
    if (tid == 0)
        atomicMax((unsigned int*)out, __float_as_uint(red[0]));
}

// quantize AND transpose: qwT[blk][f][k] = rint(w[blk][k][f]/s)*s  (f16)
__global__ void lc_quantT_kernel(const float* __restrict__ w,
                                 const float* __restrict__ absmax,
                                 _Float16* __restrict__ qwT, int n) {
    int i = blockIdx.x * 256 + threadIdx.x;
    if (i >= n) return;
    int f = i % FOUT;
    int t = i / FOUT;
    int k = t % KDIM;
    int b = t / KDIM;
    float scale = absmax[0] * (1.0f / 7.0f);
    float q = (scale > 0.0f) ? rintf(w[i] / scale) * scale : 0.0f;
    qwT[(b * FOUT + f) * KDIM + k] = (_Float16)q;
}

// ---------------- prep 3: output row -> base address table ----------------

__global__ void lc_addr_kernel(int* __restrict__ tab) {
    int m = blockIdx.x * 256 + threadIdx.x;
    if (m >= MROWS) return;
    int n  = m / (HH * WBLK);
    int r  = m - n * (HH * WBLK);
    int h  = r / WBLK;
    int wl = r - h * WBLK;
    tab[m] = ((n * FOUT) * HH + h) * WW + wl;   // blk*WBLK and f*H*W added later
}

// ---------------- main local-conv GEMM via WMMA ----------------

__global__ __launch_bounds__(256, 2)
void lc_wmma_kernel(const _Float16* __restrict__ xpad, // [N,C,58,58] f16
                    const _Float16* __restrict__ qwT,  // [SW,F,KDIM] f16
                    const int* __restrict__ addrTab,   // [MROWS]
                    const float* __restrict__ bias,    // [F]
                    float* __restrict__ out)           // [N,F,H,W] f32
{
    __shared__ __align__(16) _Float16 lA[MTILE * LDA];  // A tile: [m][k]
    __shared__ __align__(16) _Float16 lB[FOUT * LDA];   // B tile: [f][k]

    const int tid  = threadIdx.x;
    const int lane = tid & 31;
    // wave-uniform values (force into SGPRs)
    const int waveU = __builtin_amdgcn_readfirstlane(tid >> 5);
    const int wmU   = waveU & 3;      // wave M offset: 32*wmU
    const int wnU   = waveU >> 2;     // wave N offset: 64*wnU
    const int blk   = blockIdx.y;     // width block 0..3
    const int m0    = blockIdx.x * MTILE;

    // cooperative-load roles: each thread owns one row and 16 contiguous k
    const int mA   = tid & 127;                                 // A row / B col
    const int thi  = __builtin_amdgcn_readfirstlane(tid >> 7);  // wave-uniform
    const int kofs = thi * 16;

    // decode this thread's A row once (invariant over K loop)
    const int mg    = m0 + mA;
    const int nimg  = mg / (HH * WBLK);
    const int rm    = mg - nimg * (HH * WBLK);
    const int hA    = rm / WBLK;
    const int wA    = blk * WBLK + (rm - hA * WBLK);
    // padded-space base for (kh=0,kw=0) patch element of this row
    const int rowbase = nimg * (CIN * XPC) + hA * XP + wA;

    const _Float16* qrow = qwT + ((blk * FOUT + mA) * KDIM);

    v8f acc[2][4];
    const v8f vzero = {};
#pragma unroll
    for (int mi = 0; mi < 2; ++mi)
#pragma unroll
        for (int ni = 0; ni < 4; ++ni)
            acc[mi][ni] = vzero;

    for (int kt = 0; kt < KDIM / KTILE; ++kt) {
        const int kb = kt * KTILE;
        const int kgbase = kb + kofs;   // wave-uniform -> SALU offset math

        // ---- A: im2col gather, 16 contiguous k per thread ----
        v16h ta;
#pragma unroll
        for (int i = 0; i < 16; ++i) {
            const int kg = kgbase + i;          // uniform
            const int c  = kg / 9;              // SALU
            const int r  = kg - c * 9;
            const int kh = r / 3;
            const int kw = r - kh * 3;
            const int off = c * XPC + kh * XP + kw;  // uniform element offset
            ta[i] = xpad[rowbase + off];
        }
        {
            v8h lo = __builtin_shufflevector(ta, ta, 0,1,2,3,4,5,6,7);
            v8h hi = __builtin_shufflevector(ta, ta, 8,9,10,11,12,13,14,15);
            *(v8h*)&lA[mA * LDA + kofs]     = lo;
            *(v8h*)&lA[mA * LDA + kofs + 8] = hi;
        }

        // ---- B: pre-transposed weights, 2x b128 each way ----
        {
            v8h b0 = *(const v8h*)&qrow[kb + kofs];
            v8h b1 = *(const v8h*)&qrow[kb + kofs + 8];
            *(v8h*)&lB[mA * LDA + kofs]     = b0;
            *(v8h*)&lB[mA * LDA + kofs + 8] = b1;
        }
        __syncthreads();

        // ---- WMMA fragments + MACs ----
        const int half = lane >> 4;
        const int l16  = lane & 15;
#pragma unroll
        for (int mi = 0; mi < 2; ++mi) {
            const int arow = wmU * 32 + mi * 16 + l16;
            // A 16-bit layout: halves 0..7 -> K=half*8..+7 ; 8..15 -> K=16+half*8..+7
            v8h alo = *(const v8h*)&lA[arow * LDA + half * 8];
            v8h ahi = *(const v8h*)&lA[arow * LDA + 16 + half * 8];
            v16h a  = __builtin_shufflevector(alo, ahi,
                        0,1,2,3,4,5,6,7,8,9,10,11,12,13,14,15);
#pragma unroll
            for (int ni = 0; ni < 4; ++ni) {
                const int bcol = wnU * 64 + ni * 16 + l16;
                // B 16-bit layout: halves 0..7 -> K=half*8..+7 ; 8..15 -> K=16+half*8..+7
                v8h blo = *(const v8h*)&lB[bcol * LDA + half * 8];
                v8h bhi = *(const v8h*)&lB[bcol * LDA + 16 + half * 8];
                v16h b  = __builtin_shufflevector(blo, bhi,
                            0,1,2,3,4,5,6,7,8,9,10,11,12,13,14,15);
                acc[mi][ni] = __builtin_amdgcn_wmma_f32_16x16x32_f16(
                    false, a, false, b, (short)0, acc[mi][ni], false, false);
            }
        }
        __syncthreads();
    }

    // ---- epilogue: table-driven addressing (uniform loads + cndmask) ----
    const int l16    = lane & 15;
    const bool hiHalf = (lane >> 4) != 0;   // lane-variant, 2 values only

    float bv[4];
    int   fofs[4];
#pragma unroll
    for (int ni = 0; ni < 4; ++ni) {
        const int fcol = wnU * 64 + ni * 16 + l16;
        bv[ni]   = bias[fcol];
        fofs[ni] = fcol * (HH * WW);        // lane-variant f term of address
    }

    // Wave-uniform pointer into the precomputed row-address table.
    const int* tab = addrTab + (m0 + wmU * 32);
    const int blkoff = blk * WBLK;

#pragma unroll
    for (int mi = 0; mi < 2; ++mi) {
#pragma unroll
        for (int j = 0; j < 8; ++j) {
            const int s0 = tab[mi * 16 + j]     + blkoff;  // uniform load
            const int s8 = tab[mi * 16 + j + 8] + blkoff;  // uniform load
            const int sbase = hiHalf ? s8 : s0;            // one v_cndmask
#pragma unroll
            for (int ni = 0; ni < 4; ++ni)
                out[sbase + fofs[ni]] = acc[mi][ni][j] + bv[ni];
        }
    }
}

// ---------------- launch ----------------

extern "C" void kernel_launch(void* const* d_in, const int* in_sizes, int n_in,
                              void* d_out, int out_size, void* d_ws, size_t ws_size,
                              hipStream_t stream) {
    (void)in_sizes; (void)n_in; (void)out_size; (void)ws_size;

    const float* x    = (const float*)d_in[0];   // [32,64,56,56]
    const float* w    = (const float*)d_in[1];   // [1,4,576,128]
    const float* bias = (const float*)d_in[2];   // [128]
    float* out        = (float*)d_out;

    // workspace layout (~14.5 MB):
    //   [0,4)       : absmax (f32)
    //   xpad        : padded f16 input            (NPAD*2 bytes)
    //   qwT         : quantized+transposed f16 W  (SW*KDIM*FOUT*2 bytes)
    //   addrTab     : output row base addresses   (MROWS*4 bytes)
    char* ws = (char*)d_ws;
    float*    absmax  = (float*)ws;
    _Float16* xpad    = (_Float16*)(ws + 64);
    size_t    off_qwT = 64 + (size_t)NPAD * 2;
    _Float16* qwT     = (_Float16*)(ws + off_qwT);
    size_t    off_tab = (off_qwT + (size_t)SW * KDIM * FOUT * 2 + 127) & ~(size_t)127;
    int*      addrTab = (int*)(ws + off_tab);
    const int nw = SW * KDIM * FOUT;  // 294912

    hipMemsetAsync(d_ws, 0, 64, stream);
    lc_pad_kernel<<<(NPAD + 255) / 256, 256, 0, stream>>>(x, xpad);
    lc_absmax_kernel<<<64, 256, 0, stream>>>(w, absmax, nw);
    lc_quantT_kernel<<<(nw + 255) / 256, 256, 0, stream>>>(w, absmax, qwT, nw);
    lc_addr_kernel<<<(MROWS + 255) / 256, 256, 0, stream>>>(addrTab);

    dim3 grid(MROWS / MTILE, SW);    // 196 x 4
    lc_wmma_kernel<<<grid, 256, 0, stream>>>(xpad, qwT, addrTab, bias, out);
}
